// DOGS_23699629539852
// MI455X (gfx1250) — compile-verified, matrix operand order checked by reading, not run on hardware
//
#include <hip/hip_runtime.h>
#include <hip/hip_bf16.h>

// ---------------- problem constants ----------------
#define BDIM  512
#define SQ_   49
#define SK_   50
#define DIMG_ 2048
#define DTXT_ 768
#define DM    512
#define NH    4
#define DH_   128

typedef __attribute__((ext_vector_type(16))) _Float16 v16h;
typedef __attribute__((ext_vector_type(8)))  _Float16 v8h;
typedef __attribute__((ext_vector_type(8)))  float    v8f;
typedef __attribute__((ext_vector_type(4)))  unsigned int u32x4;
typedef __attribute__((ext_vector_type(8)))  int          i32x8;
typedef __attribute__((ext_vector_type(4)))  int          i32x4;

// ---------------------------------------------------------------------------
// WMMA fragment loader (wave32, 16-bit A/B layout):
//   lanes 0-15 : row = row0+lane,    K = {k0..k0+7, k0+16..k0+23}
//   lanes16-31 : row = row0+lane-16, K = {k0+8..k0+15, k0+24..k0+31}
// tile row-major f16 in LDS, leading dim `ld` (multiple of 8 halves).
// ---------------------------------------------------------------------------
__device__ __forceinline__ v16h ld_frag(const _Float16* tile, int row0, int k0, int ld) {
  const int lane = threadIdx.x & 31;
  const int sub  = lane & 15;
  const int hi   = lane >> 4;
  const _Float16* p = tile + (size_t)(row0 + sub) * ld + k0 + hi * 8;
  const v8h* vp = reinterpret_cast<const v8h*>(p);
  v8h lo = vp[0];
  v8h hh = vp[2];  // +16 halves
  return __builtin_shufflevector(lo, hh, 0,1,2,3,4,5,6,7,8,9,10,11,12,13,14,15);
}

__device__ __forceinline__ v8f wmma_f16(v16h a, v16h b, v8f c) {
  return __builtin_amdgcn_wmma_f32_16x16x32_f16(false, a, false, b, (short)0, c, false, false);
}

// ---------------------------------------------------------------------------
// TDM: 2-D f16 tile load Global -> LDS with LDS row padding.
// D# group0: count=1, lds_addr, global tile address, type=2.
// D# group1: data_size=2B, pad fields, tensor dims/strides, tile dims.
// pad_interval code: DWORDs before pad = 2<<code ; pad_amount code: DWORDs = code+1
// 6-arg builtin (clang-23): (g0, g1, g2, g3, g4, cpol)
// ---------------------------------------------------------------------------
__device__ __forceinline__ void tdm_load_2d(unsigned lds_off, const _Float16* gaddr,
                                            unsigned tensor_d0, unsigned tensor_d1,
                                            unsigned tile_d0, unsigned tile_d1,
                                            unsigned long long stride0_elems,
                                            unsigned pad_interval_code,
                                            unsigned pad_amount_code, int pad_en)
{
  unsigned long long ga = (unsigned long long)(uintptr_t)gaddr;
  u32x4 g0;
  g0[0] = 1u;                                              // count=1 (valid user D#)
  g0[1] = lds_off;                                         // lds_addr [63:32]
  g0[2] = (unsigned)(ga & 0xffffffffu);                    // global_addr lo
  g0[3] = (unsigned)((ga >> 32) & 0x01ffffffu) | (2u << 30); // addr hi + type=2
  i32x8 g1;
  unsigned w0 = (1u << 16);                                // data_size = 2 bytes
  if (pad_en) w0 |= (1u << 20) | (pad_interval_code << 22) | (pad_amount_code << 25);
  g1[0] = (int)w0;                                         // wg_mask=0 | flags
  g1[1] = (int)((tensor_d0 & 0xffffu) << 16);              // abar=0 | d0.lo16
  g1[2] = (int)(((tensor_d0 >> 16) & 0xffffu) | ((tensor_d1 & 0xffffu) << 16));
  g1[3] = (int)(((tensor_d1 >> 16) & 0xffffu) | ((tile_d0 & 0xffffu) << 16));
  g1[4] = (int)(tile_d1 & 0xffffu);                        // tile_dim1 | tile_dim2=0
  g1[5] = (int)((unsigned)(stride0_elems & 0xffffffffu));  // dim0 stride lo32
  g1[6] = (int)((unsigned)((stride0_elems >> 32) & 0xffffu)); // hi16 | dim1stride.lo=0
  g1[7] = 0;
  i32x4 z4 = {0, 0, 0, 0};
  i32x8 z8 = {0, 0, 0, 0, 0, 0, 0, 0};
  __builtin_amdgcn_tensor_load_to_lds(g0, g1, z4, z4, z8, 0);
}

__device__ __forceinline__ unsigned lds_off_u32(const void* p) {
  return (unsigned)(unsigned long long)(uintptr_t)p;  // low 32 bits = LDS offset
}

// ---------------------------------------------------------------------------
// TDM-fed, double-buffered WMMA GEMM: C[M,N] = A[M,K] @ Bt[N,K]^T (+ epilogue)
// A  : f16 row-major MxK        Bt : f16 row-major NxK (pre-transposed B)
// M % 128 == 0, N % 128 == 0, K % 32 == 0 (true for all nine GEMMs here).
//   EPI 0: Cf = acc+bias
//   EPI 2: Cf = p = acc+bias ; Ch = gelu(p)
//   EPI 3: Cf = acc+bias+resid (resid may alias Cf; one thread per element)
//   EPI 4: Ch = acc+bias (f16 only)
// block = 256 threads (8 waves), tile 128x128, BK=32, LDS 40KB (2x ping-pong)
// ---------------------------------------------------------------------------
template<int EPI>
__global__ __launch_bounds__(256)
void gemm_wmma(const _Float16* __restrict__ A, const _Float16* __restrict__ Bt,
               const float* __restrict__ bias, const float* __restrict__ resid,
               float* __restrict__ Cf, _Float16* __restrict__ Ch,
               int M, int N, int K)
{
  constexpr int BM = 128, BN = 128, BK = 32, LDT = BK + 8;   // 40-half rows
  __shared__ __align__(16) _Float16 As[2][BM * LDT];
  __shared__ __align__(16) _Float16 Bs[2][BN * LDT];

  const int tid  = threadIdx.x;
  const int wave = tid >> 5;
  const int wm   = wave >> 1;          // 0..3 -> 32-row slab
  const int wn   = wave & 1;           // 0..1 -> 64-col slab
  const int bm0  = blockIdx.y * BM;
  const int bn0  = blockIdx.x * BN;

  v8f acc[2][4];
  const v8f vzero = {0.f,0.f,0.f,0.f,0.f,0.f,0.f,0.f};
  for (int i = 0; i < 2; ++i)
    for (int j = 0; j < 4; ++j) acc[i][j] = vzero;

  const int nk = K / BK;
  // TDM issue: tile rows = 128, row length = 32 halves (64B), LDS rows padded
  // to 40 halves via pad fields (16 DWORDs data -> code 3, 4 DWORDs pad -> code 3).
  auto issue = [&](int kt, int buf) {
    const _Float16* ga = A  + (size_t)bm0 * K + kt * BK;
    const _Float16* gb = Bt + (size_t)bn0 * K + kt * BK;
    tdm_load_2d(lds_off_u32(&As[buf][0]), ga,
                (unsigned)(K - kt * BK), (unsigned)BM, BK, BM,
                (unsigned long long)K, 3u, 3u, 1);
    tdm_load_2d(lds_off_u32(&Bs[buf][0]), gb,
                (unsigned)(K - kt * BK), (unsigned)BN, BK, BN,
                (unsigned long long)K, 3u, 3u, 1);
  };

  if (wave == 0) issue(0, 0);
  for (int kt = 0; kt < nk; ++kt) {
    const int buf = kt & 1;
    if (wave == 0) {
      if (kt + 1 < nk) {
        issue(kt + 1, buf ^ 1);                       // prefetch next tile
        __builtin_amdgcn_s_wait_tensorcnt(2);         // current tile landed
      } else {
        __builtin_amdgcn_s_wait_tensorcnt(0);
      }
    }
    __syncthreads();                                  // tile visible to all waves
    const _Float16* as = As[buf];
    const _Float16* bs = Bs[buf];
    v16h afr0 = ld_frag(as, wm * 32 + 0,  0, LDT);
    v16h afr1 = ld_frag(as, wm * 32 + 16, 0, LDT);
    for (int j = 0; j < 4; ++j) {
      v16h bfr = ld_frag(bs, wn * 64 + j * 16, 0, LDT);
      acc[0][j] = wmma_f16(afr0, bfr, acc[0][j]);
      acc[1][j] = wmma_f16(afr1, bfr, acc[1][j]);
    }
    __syncthreads();                                  // done reading before reuse
  }

  // epilogue (C layout: lane&15 = n, vgpr r -> m = r + (lane>>4)*8)
  const int lane = tid & 31;
  const int sub = lane & 15, hi = lane >> 4;
  for (int mi = 0; mi < 2; ++mi)
    for (int j = 0; j < 4; ++j)
      for (int r = 0; r < 8; ++r) {
        int m = bm0 + wm * 32 + mi * 16 + r + hi * 8;
        int n = bn0 + wn * 64 + j * 16 + sub;
        if (m >= M) continue;
        float v = acc[mi][j][r];
        if (bias) v += bias[n];
        size_t off = (size_t)m * N + n;
        if (EPI == 0) {
          Cf[off] = v;
        } else if (EPI == 2) {
          Cf[off] = v;
          float g = 0.5f * v * (1.0f + erff(v * 0.70710678118654752f));
          Ch[off] = (_Float16)g;
        } else if (EPI == 3) {
          v += resid[off];
          Cf[off] = v;
        } else if (EPI == 4) {
          Ch[off] = (_Float16)v;
        }
      }
}

// ---------------------------------------------------------------------------
// prep kernels: f32 -> f16 conversions / transposes (one-time, memory bound)
// ---------------------------------------------------------------------------
__global__ __launch_bounds__(256)
void cvt_f16(const float* __restrict__ in, _Float16* __restrict__ out, size_t n)
{
  size_t i = (size_t)blockIdx.x * 256 + threadIdx.x;
  if (i < n) out[i] = (_Float16)in[i];
}

// Bt[n*K + k] = W[k*N + n]  (weights pre-transposed for TDM B tiles)
__global__ __launch_bounds__(256)
void wT_f16(const float* __restrict__ W, _Float16* __restrict__ Bt, int K, int N)
{
  size_t i = (size_t)blockIdx.x * 256 + threadIdx.x;
  if (i >= (size_t)K * N) return;
  int n = (int)(i % N);
  int k = (int)(i / N);
  Bt[(size_t)n * K + k] = (_Float16)W[i];
}

// img_h[(b*SQ+s)*DIMG + c] = img[(b*DIMG+c)*SQ + s]
__global__ __launch_bounds__(256)
void imgT_f16(const float* __restrict__ img, _Float16* __restrict__ out)
{
  size_t i = (size_t)blockIdx.x * 256 + threadIdx.x;
  if (i >= (size_t)BDIM * DIMG_ * SQ_) return;
  int s = (int)(i % SQ_);
  size_t t = i / SQ_;
  int c = (int)(t % DIMG_);
  int b = (int)(t / DIMG_);
  out[((size_t)(b * SQ_ + s)) * DIMG_ + c] = (_Float16)img[i];
}

// ---------------------------------------------------------------------------
// Row LayerNorm over D=512: Of = g*(h-mu)*rsqrt(var+eps)+be ; Oh = f16 copy
// ---------------------------------------------------------------------------
__global__ __launch_bounds__(256)
void ln_kernel(const float* __restrict__ Hin, const float* __restrict__ g,
               const float* __restrict__ be, float* __restrict__ Of,
               _Float16* __restrict__ Oh)
{
  __shared__ float red[256];
  const int row = blockIdx.x, tid = threadIdx.x;
  const float* hr = Hin + (size_t)row * DM;
  float x0 = hr[tid], x1 = hr[tid + 256];
  red[tid] = x0 + x1; __syncthreads();
  for (int o = 128; o > 0; o >>= 1) { if (tid < o) red[tid] += red[tid + o]; __syncthreads(); }
  float mu = red[0] * (1.0f / DM);
  __syncthreads();
  float d0 = x0 - mu, d1 = x1 - mu;
  red[tid] = d0 * d0 + d1 * d1; __syncthreads();
  for (int o = 128; o > 0; o >>= 1) { if (tid < o) red[tid] += red[tid + o]; __syncthreads(); }
  float inv = rsqrtf(red[0] * (1.0f / DM) + 1e-5f);
  float y0 = g[tid] * d0 * inv + be[tid];
  float y1 = g[tid + 256] * d1 * inv + be[tid + 256];
  size_t base = (size_t)row * DM;
  Of[base + tid] = y0;           Of[base + tid + 256] = y1;
  Oh[base + tid] = (_Float16)y0; Oh[base + tid + 256] = (_Float16)y1;
}

// ---------------------------------------------------------------------------
// Fused per-(b,h) attention: scores = q@k^T*scale -> softmax -> o = a@v
// LDS overlays: sc(f32) over qs, af(f16) over ks (phase-separated by barriers)
// ---------------------------------------------------------------------------
#define ATTN_SMEM 53248
__global__ __launch_bounds__(256)
void attn_kernel(const _Float16* __restrict__ q, const _Float16* __restrict__ k,
                 const _Float16* __restrict__ v, _Float16* __restrict__ o,
                 float* __restrict__ abuf)
{
  extern __shared__ __align__(16) char smem[];
  _Float16* qs  = (_Float16*)(smem);            // 64*136 f16 = 17408 B
  _Float16* ks  = (_Float16*)(smem + 17408);    // 64*136 f16 = 17408 B
  _Float16* vts = (_Float16*)(smem + 34816);    // 128*72 f16 = 18432 B  [d][sk]
  float*    sc  = (float*)(smem);               // 64*65 f32 (overlays qs)
  _Float16* af  = (_Float16*)(smem + 17408);    // 64*72 f16 (overlays ks)

  const int bh = blockIdx.x, b = bh >> 2, h = bh & 3;
  const int tid = threadIdx.x, wave = tid >> 5;
  const int sub = tid & 15, hi = (tid >> 4) & 1;

  for (int i = 0; i < 32; ++i) {                // stage q/k rows (pad -> 0)
    int e = i * 256 + tid;
    int c = e & 127, r = e >> 7;
    _Float16 qv = (_Float16)0, kv = (_Float16)0;
    if (r < SQ_) qv = q[((size_t)(b * SQ_ + r)) * DM + h * DH_ + c];
    if (r < SK_) kv = k[((size_t)(b * SK_ + r)) * DM + h * DH_ + c];
    qs[r * 136 + c] = qv;
    ks[r * 136 + c] = kv;
  }
  for (int i = 0; i < 32; ++i) {                // stage v transposed [d][sk]
    int e = i * 256 + tid;
    int skk = e & 63, d = e >> 6;
    _Float16 vv = (_Float16)0;
    if (skk < SK_) vv = v[((size_t)(b * SK_ + skk)) * DM + h * DH_ + d];
    vts[d * 72 + skk] = vv;
  }
  __syncthreads();

  // scores: 16 fragments (4m x 4n), 2 per wave; K = 128 -> 4 steps
  v8f sacc[2];
  for (int fi = 0; fi < 2; ++fi) {
    int f = wave * 2 + fi;
    int mf = f >> 2, nf = f & 3;
    v8f a = {0.f,0.f,0.f,0.f,0.f,0.f,0.f,0.f};
    for (int kk = 0; kk < 4; ++kk) {
      v16h qa = ld_frag(qs, mf * 16, kk * 32, 136);
      v16h kb = ld_frag(ks, nf * 16, kk * 32, 136);
      a = wmma_f16(qa, kb, a);
    }
    sacc[fi] = a;
  }
  __syncthreads();                              // all reads of qs/ks complete
  for (int fi = 0; fi < 2; ++fi) {
    int f = wave * 2 + fi;
    int mf = f >> 2, nf = f & 3;
    for (int r = 0; r < 8; ++r) {
      int m = mf * 16 + r + hi * 8, n = nf * 16 + sub;
      sc[m * 65 + n] = sacc[fi][r] * 0.08838834764831845f;  // 1/sqrt(128)
    }
  }
  __syncthreads();

  // softmax: one row per thread; write a (f16, padded 0) + per-head weights
  if (tid < 64) {
    if (tid < SQ_) {
      float mx = -1e30f;
      for (int n = 0; n < SK_; ++n) mx = fmaxf(mx, sc[tid * 65 + n]);
      float sum = 0.f;
      for (int n = 0; n < SK_; ++n) {
        float e = __expf(sc[tid * 65 + n] - mx);
        sc[tid * 65 + n] = e;
        sum += e;
      }
      float inv = 1.f / sum;
      size_t abase = (((size_t)(b * NH + h)) * SQ_ + tid) * SK_;
      for (int n = 0; n < 64; ++n) {
        float a = (n < SK_) ? sc[tid * 65 + n] * inv : 0.f;
        af[tid * 72 + n] = (_Float16)a;
        if (n < SK_) abuf[abase + n] = a;
      }
    } else {
      for (int n = 0; n < 64; ++n) af[tid * 72 + n] = (_Float16)0;
    }
  }
  __syncthreads();

  // o = a @ v : 32 fragments (4m x 8n), 4 per wave; K = 64 -> 2 steps
  for (int fi = 0; fi < 4; ++fi) {
    int f = wave * 4 + fi;
    int mf = f >> 3, nf = f & 7;
    v8f a = {0.f,0.f,0.f,0.f,0.f,0.f,0.f,0.f};
    for (int kk = 0; kk < 2; ++kk) {
      v16h aa = ld_frag(af,  mf * 16, kk * 32, 72);
      v16h vb = ld_frag(vts, nf * 16, kk * 32, 72);
      a = wmma_f16(aa, vb, a);
    }
    for (int r = 0; r < 8; ++r) {
      int m = mf * 16 + r + hi * 8, n = nf * 16 + sub;
      if (m < SQ_)
        o[((size_t)(b * SQ_ + m)) * DM + h * DH_ + n] = (_Float16)a[r];
    }
  }
}

// attn_weights = mean over heads
__global__ __launch_bounds__(256)
void aw_mean(const float* __restrict__ abuf, float* __restrict__ out)
{
  int idx = blockIdx.x * 256 + threadIdx.x;
  if (idx >= BDIM * SQ_ * SK_) return;
  int b = idx / (SQ_ * SK_);
  int rem = idx - b * (SQ_ * SK_);
  float s = 0.f;
  for (int h = 0; h < NH; ++h)
    s += abuf[((size_t)(b * NH + h)) * (SQ_ * SK_) + rem];
  out[idx] = s * 0.25f;
}

// n1 = normalize(mean over SQ of attn_output) -> f16
__global__ __launch_bounds__(256)
void n1_kernel(const float* __restrict__ attn, _Float16* __restrict__ n1)
{
  __shared__ float red[256];
  const int b = blockIdx.x, tid = threadIdx.x;
  float s0 = 0.f, s1 = 0.f;
  for (int qq = 0; qq < SQ_; ++qq) {
    size_t base = ((size_t)(b * SQ_ + qq)) * DM;
    s0 += attn[base + tid];
    s1 += attn[base + tid + 256];
  }
  s0 *= (1.0f / SQ_); s1 *= (1.0f / SQ_);
  red[tid] = s0 * s0 + s1 * s1; __syncthreads();
  for (int o = 128; o > 0; o >>= 1) { if (tid < o) red[tid] += red[tid + o]; __syncthreads(); }
  float inv = rsqrtf(red[0]);
  n1[(size_t)b * DM + tid]       = (_Float16)(s0 * inv);
  n1[(size_t)b * DM + tid + 256] = (_Float16)(s1 * inv);
}

// n2 = normalize(pt[:,0,:]) -> f16 row-major (acts as pre-transposed B = n2^T)
__global__ __launch_bounds__(256)
void n2_kernel(const float* __restrict__ pt, _Float16* __restrict__ n2h)
{
  __shared__ float red[256];
  const int b = blockIdx.x, tid = threadIdx.x;
  const float* r = pt + (size_t)b * SK_ * DM;   // row 0 of pt[b]
  float x0 = r[tid], x1 = r[tid + 256];
  red[tid] = x0 * x0 + x1 * x1; __syncthreads();
  for (int o = 128; o > 0; o >>= 1) { if (tid < o) red[tid] += red[tid + o]; __syncthreads(); }
  float inv = rsqrtf(red[0]);
  n2h[(size_t)b * DM + tid]       = (_Float16)(x0 * inv);
  n2h[(size_t)b * DM + tid + 256] = (_Float16)(x1 * inv);
}

// ---------------------------------------------------------------------------
extern "C" void kernel_launch(void* const* d_in, const int* in_sizes, int n_in,
                              void* d_out, int out_size, void* d_ws, size_t ws_size,
                              hipStream_t stream) {
  (void)in_sizes; (void)n_in; (void)out_size; (void)ws_size;
  const float* img = (const float*)d_in[0];
  const float* txt = (const float*)d_in[1];
  const float* Wi1 = (const float*)d_in[2];  const float* bi1 = (const float*)d_in[3];
  const float* Wi2 = (const float*)d_in[4];  const float* bi2 = (const float*)d_in[5];
  const float* gi  = (const float*)d_in[6];  const float* bei = (const float*)d_in[7];
  const float* Wt1 = (const float*)d_in[8];  const float* bt1 = (const float*)d_in[9];
  const float* Wt2 = (const float*)d_in[10]; const float* bt2 = (const float*)d_in[11];
  const float* gt  = (const float*)d_in[12]; const float* bet = (const float*)d_in[13];
  const float* Wq  = (const float*)d_in[14]; const float* bq  = (const float*)d_in[15];
  const float* Wk  = (const float*)d_in[16]; const float* bk  = (const float*)d_in[17];
  const float* Wv  = (const float*)d_in[18]; const float* bv  = (const float*)d_in[19];
  const float* Wo  = (const float*)d_in[20]; const float* bo  = (const float*)d_in[21];

  float* out = (float*)d_out;
  const size_t MI = (size_t)BDIM * SQ_;   // 25088 = 196*128
  const size_t MT = (size_t)BDIM * SK_;   // 25600 = 200*128
  const size_t o_score = 0;
  const size_t o_attn  = (size_t)BDIM * BDIM;
  const size_t o_aw    = o_attn + MI * DM;
  const size_t o_pi    = o_aw + (size_t)BDIM * SQ_ * SK_;
  const size_t o_pt    = o_pi + MI * DM;

  // workspace carve
  char* ws = (char*)d_ws;
  size_t off = 0;
  auto carve = [&](size_t bytes) -> char* {
    char* p = ws + off; off += (bytes + 255) & ~(size_t)255; return p;
  };
  _Float16* img_h = (_Float16*)carve(MI * DIMG_ * 2);         // transposed image, f16
  _Float16* txt_h = (_Float16*)carve(MT * DTXT_ * 2);
  _Float16* Wi1t  = (_Float16*)carve((size_t)DIMG_ * DM * 2); // weights N x K f16
  _Float16* Wi2t  = (_Float16*)carve((size_t)DM * DM * 2);
  _Float16* Wt1t  = (_Float16*)carve((size_t)DTXT_ * DM * 2);
  _Float16* Wt2t  = (_Float16*)carve((size_t)DM * DM * 2);
  _Float16* Wqt   = (_Float16*)carve((size_t)DM * DM * 2);
  _Float16* Wkt   = (_Float16*)carve((size_t)DM * DM * 2);
  _Float16* Wvt   = (_Float16*)carve((size_t)DM * DM * 2);
  _Float16* Wot   = (_Float16*)carve((size_t)DM * DM * 2);
  float*    p_img = (float*)   carve(MI * DM * 4);
  _Float16* g_img = (_Float16*)carve(MI * DM * 2);
  _Float16* pi_h  = (_Float16*)carve(MI * DM * 2);
  float*    p_txt = (float*)   carve(MT * DM * 4);
  _Float16* g_txt = (_Float16*)carve(MT * DM * 2);
  _Float16* pt_h  = (_Float16*)carve(MT * DM * 2);
  _Float16* q_h   = (_Float16*)carve(MI * DM * 2);
  _Float16* k_h   = (_Float16*)carve(MT * DM * 2);
  _Float16* v_h   = (_Float16*)carve(MT * DM * 2);
  _Float16* o_h   = (_Float16*)carve(MI * DM * 2);
  float*    abuf  = (float*)   carve((size_t)BDIM * NH * SQ_ * SK_ * 4);
  _Float16* n1h   = (_Float16*)carve((size_t)BDIM * DM * 2);
  _Float16* n2h   = (_Float16*)carve((size_t)BDIM * DM * 2);

  dim3 blk(256);
  auto gg = [](size_t M, int N) { return dim3((unsigned)((N + 127) / 128), (unsigned)((M + 127) / 128)); };
  auto eg = [](size_t n) { return dim3((unsigned)((n + 255) / 256)); };

  // ---- prep: f16 conversions / transposes ----
  imgT_f16<<<eg(MI * DIMG_), blk, 0, stream>>>(img, img_h);
  cvt_f16 <<<eg(MT * DTXT_), blk, 0, stream>>>(txt, txt_h, MT * DTXT_);
  wT_f16  <<<eg((size_t)DIMG_ * DM), blk, 0, stream>>>(Wi1, Wi1t, DIMG_, DM);
  wT_f16  <<<eg((size_t)DM * DM),    blk, 0, stream>>>(Wi2, Wi2t, DM, DM);
  wT_f16  <<<eg((size_t)DTXT_ * DM), blk, 0, stream>>>(Wt1, Wt1t, DTXT_, DM);
  wT_f16  <<<eg((size_t)DM * DM),    blk, 0, stream>>>(Wt2, Wt2t, DM, DM);
  wT_f16  <<<eg((size_t)DM * DM),    blk, 0, stream>>>(Wq,  Wqt,  DM, DM);
  wT_f16  <<<eg((size_t)DM * DM),    blk, 0, stream>>>(Wk,  Wkt,  DM, DM);
  wT_f16  <<<eg((size_t)DM * DM),    blk, 0, stream>>>(Wv,  Wvt,  DM, DM);
  wT_f16  <<<eg((size_t)DM * DM),    blk, 0, stream>>>(Wo,  Wot,  DM, DM);

  // ---- image projection head ----
  gemm_wmma<2><<<gg(MI, DM), blk, 0, stream>>>(img_h, Wi1t, bi1, nullptr, p_img, g_img, (int)MI, DM, DIMG_);
  gemm_wmma<3><<<gg(MI, DM), blk, 0, stream>>>(g_img, Wi2t, bi2, p_img,   p_img, nullptr,(int)MI, DM, DM);
  ln_kernel<<<dim3((unsigned)MI), blk, 0, stream>>>(p_img, gi, bei, out + o_pi, pi_h);
  // ---- text projection head ----
  gemm_wmma<2><<<gg(MT, DM), blk, 0, stream>>>(txt_h, Wt1t, bt1, nullptr, p_txt, g_txt, (int)MT, DM, DTXT_);
  gemm_wmma<3><<<gg(MT, DM), blk, 0, stream>>>(g_txt, Wt2t, bt2, p_txt,   p_txt, nullptr,(int)MT, DM, DM);
  ln_kernel<<<dim3((unsigned)MT), blk, 0, stream>>>(p_txt, gt, bet, out + o_pt, pt_h);
  // ---- q/k/v projections (f16 out) ----
  gemm_wmma<4><<<gg(MI, DM), blk, 0, stream>>>(pi_h, Wqt, bq, nullptr, nullptr, q_h, (int)MI, DM, DM);
  gemm_wmma<4><<<gg(MT, DM), blk, 0, stream>>>(pt_h, Wkt, bk, nullptr, nullptr, k_h, (int)MT, DM, DM);
  gemm_wmma<4><<<gg(MT, DM), blk, 0, stream>>>(pt_h, Wvt, bv, nullptr, nullptr, v_h, (int)MT, DM, DM);
  // ---- fused attention per (b,h) ----
  attn_kernel<<<dim3(BDIM * NH), blk, ATTN_SMEM, stream>>>(q_h, k_h, v_h, o_h, abuf);
  aw_mean<<<eg((size_t)BDIM * SQ_ * SK_), blk, 0, stream>>>(abuf, out + o_aw);
  // ---- output projection ----
  gemm_wmma<0><<<gg(MI, DM), blk, 0, stream>>>(o_h, Wot, bo, nullptr, out + o_attn, nullptr, (int)MI, DM, DM);
  // ---- cosine-similarity score ----
  n1_kernel<<<dim3(BDIM), blk, 0, stream>>>(out + o_attn, n1h);
  n2_kernel<<<dim3(BDIM), blk, 0, stream>>>(out + o_pt, n2h);
  gemm_wmma<0><<<gg(BDIM, BDIM), blk, 0, stream>>>(n1h, n2h, nullptr, nullptr, out + o_score, nullptr, BDIM, BDIM, DM);
}
